// GINAutoencoder_48163763257711
// MI455X (gfx1250) — matrix operand order repaired and checked by reading, not compiled, not run on hardware
//
#include <hip/hip_runtime.h>

typedef __attribute__((ext_vector_type(16))) _Float16 v16h;
typedef __attribute__((ext_vector_type(8)))  _Float16 v8h;
typedef __attribute__((ext_vector_type(8)))  float    v8f;

#define NND   100000
#define NED   1600000
#define FEATS 128
#define OUTF  64
#define NGR   64

__global__ void k_zero(float* __restrict__ p, int n) {
  int t = blockIdx.x * blockDim.x + threadIdx.x;
  if (t < n) p[t] = 0.0f;
}

// transpose + convert weights: wt[n*K + k] = (f16) w[k*N + n]
__global__ void k_convw(const float* __restrict__ w, _Float16* __restrict__ wt, int K, int N) {
  int t = blockIdx.x * blockDim.x + threadIdx.x;
  if (t < K * N) {
    int k = t / N, n = t % N;
    wt[(size_t)n * K + k] = (_Float16)w[t];
  }
}

__global__ void k_degree(const int* __restrict__ dst, float* __restrict__ deg, int E) {
  int e = blockIdx.x * blockDim.x + threadIdx.x;
  if (e < E) atomicAdd(&deg[dst[e]], 1.0f);
}

// 32 threads/edge, float4 per thread: agg[dst] += x[src]
__global__ void k_scatter(const float* __restrict__ x, const int* __restrict__ src,
                          const int* __restrict__ dst, float* __restrict__ agg, int E) {
  int t = blockIdx.x * blockDim.x + threadIdx.x;
  int e = t >> 5, q = t & 31;
  if (e >= E) return;
  int s = src[e], d = dst[e];
  float4 v = reinterpret_cast<const float4*>(x + (size_t)s * FEATS)[q];
  float* a = agg + (size_t)d * FEATS + q * 4;
  atomicAdd(a + 0, v.x); atomicAdd(a + 1, v.y);
  atomicAdd(a + 2, v.z); atomicAdd(a + 3, v.w);
}

// aH[i,k] = (f16)( x[i,k] + (deg>0 ? agg[i,k]/deg : 0) )
__global__ void k_prep(const float* __restrict__ x, const float* __restrict__ agg,
                       const float* __restrict__ deg, _Float16* __restrict__ out) {
  int t = blockIdx.x * blockDim.x + threadIdx.x;
  int i = t >> 5, q = t & 31;
  if (i >= NND) return;
  float dg = deg[i];
  float inv = dg > 0.0f ? 1.0f / dg : 0.0f;
  float4 xv = reinterpret_cast<const float4*>(x   + (size_t)i * FEATS)[q];
  float4 av = reinterpret_cast<const float4*>(agg + (size_t)i * FEATS)[q];
  _Float16* o = out + (size_t)i * FEATS + q * 4;
  o[0] = (_Float16)(xv.x + av.x * inv);
  o[1] = (_Float16)(xv.y + av.y * inv);
  o[2] = (_Float16)(xv.z + av.z * inv);
  o[3] = (_Float16)(xv.w + av.w * inv);
}

__global__ void k_count(const int* __restrict__ gid, float* __restrict__ cnt) {
  int t = blockIdx.x * blockDim.x + threadIdx.x;
  if (t < NND) atomicAdd(&cnt[gid[t]], 1.0f);
}

__global__ void k_finalize(const float* __restrict__ pool, const float* __restrict__ cnt,
                           float* __restrict__ hg, _Float16* __restrict__ hgH) {
  int t = blockIdx.x * blockDim.x + threadIdx.x;
  if (t < NGR * FEATS) {
    int g = t >> 7;
    float v = pool[t] / fmaxf(cnt[g], 1.0f);
    hg[t]  = v;
    hgH[t] = (_Float16)v;
  }
}

// One wave per 16x16 output tile row-block; A:[M,K] f16 row-major, Wt:[N,K] f16
// (pre-transposed). Fragments built per CDNA5 ISA layouts (CDNA5 ISA §7.12.2).
template <bool RELU, bool POOL, bool OUTH, int N, int K>
__global__ void k_gemm(const _Float16* __restrict__ A, const _Float16* __restrict__ Wt,
                       const float* __restrict__ bias, float* __restrict__ outF,
                       _Float16* __restrict__ outHp, const int* __restrict__ gid,
                       float* __restrict__ pool, int M) {
  const int wave = (blockIdx.x * blockDim.x + threadIdx.x) >> 5;
  const int mTiles = M >> 4;
  if (wave >= mTiles) return;                 // uniform per wave: EXEC stays all-1s
  const int lane = threadIdx.x & 31;
  const int hf   = lane >> 4;                 // lane half (0 or 1)
  const int l16  = lane & 15;
  const int mb   = wave << 4;
  constexpr int NT = N / 16;
  constexpr int KS = K / 32;

  v8f acc[NT];
  v8f z = {};
#pragma unroll
  for (int nt = 0; nt < NT; ++nt) acc[nt] = z;

  // A fragment base: row (mb + l16), K-offset hf*8 within each 32-wide k-step
  const _Float16* arow = A + (size_t)(mb + l16) * K + hf * 8;

#pragma unroll
  for (int ks = 0; ks < KS; ++ks) {
    v8h alo = *reinterpret_cast<const v8h*>(arow + ks * 32);        // K = ks*32 + hf*8 + 0..7
    v8h ahi = *reinterpret_cast<const v8h*>(arow + ks * 32 + 16);   // K = ks*32 + 16 + hf*8 + 0..7
    v16h a = __builtin_shufflevector(alo, ahi, 0,1,2,3,4,5,6,7,8,9,10,11,12,13,14,15);
#pragma unroll
    for (int nt = 0; nt < NT; ++nt) {
      // B fragment: n = nt*16 + l16, K = ks*32 + hf*8 + 0..7 and ks*32 + 16 + hf*8 + 0..7
      const _Float16* wrow = Wt + (size_t)(nt * 16 + l16) * K + ks * 32 + hf * 8;
      v8h blo = *reinterpret_cast<const v8h*>(wrow);
      v8h bhi = *reinterpret_cast<const v8h*>(wrow + 16);
      v16h b = __builtin_shufflevector(blo, bhi, 0,1,2,3,4,5,6,7,8,9,10,11,12,13,14,15);
      acc[nt] = __builtin_amdgcn_wmma_f32_16x16x32_f16(
          false, a, false, b, (short)0, acc[nt], false, false);
    }
  }

  // C/D layout: lane half selects M in {r, 8+r}; n = l16 within the 16-wide tile
  const int mrow0 = mb + hf * 8;
#pragma unroll
  for (int nt = 0; nt < NT; ++nt) {
    int n = nt * 16 + l16;
    float bn = bias[n];
#pragma unroll
    for (int r = 0; r < 8; ++r) {
      float v = acc[nt][r] + bn;
      if (RELU) v = v > 0.0f ? v : 0.0f;
      int m = mrow0 + r;
      if (POOL) {
        atomicAdd(&pool[(size_t)gid[m] * FEATS + n], v);
      } else if (OUTH) {
        outHp[(size_t)m * N + n] = (_Float16)v;
      } else {
        outF[(size_t)m * N + n] = v;
      }
    }
  }
}

extern "C" void kernel_launch(void* const* d_in, const int* in_sizes, int n_in,
                              void* d_out, int out_size, void* d_ws, size_t ws_size,
                              hipStream_t stream) {
  const float* feat = (const float*)d_in[0];
  const int*   src  = (const int*)d_in[1];
  const int*   dst  = (const int*)d_in[2];
  const int*   gid  = (const int*)d_in[3];
  const float* W1   = (const float*)d_in[4];
  const float* b1   = (const float*)d_in[5];
  const float* W2   = (const float*)d_in[6];
  const float* b2   = (const float*)d_in[7];
  const float* Wd1  = (const float*)d_in[8];
  const float* bd1  = (const float*)d_in[9];
  const float* Wd2  = (const float*)d_in[10];
  const float* bd2  = (const float*)d_in[11];
  float* out = (float*)d_out;

  // Workspace layout (256B-aligned carve-out, ~129 MB total)
  char* ws = (char*)d_ws;
  size_t off = 0;
  auto carve = [&](size_t bytes) -> char* {
    char* p = ws + off;
    off = (off + bytes + 255) & ~(size_t)255;
    return p;
  };
  float*    deg  = (float*)   carve((size_t)NND * 4);
  float*    agg  = (float*)   carve((size_t)NND * FEATS * 4);
  _Float16* aH   = (_Float16*)carve((size_t)NND * FEATS * 2);
  float*    h1   = (float*)   carve((size_t)NND * FEATS * 4);
  _Float16* w1h  = (_Float16*)carve((size_t)FEATS * FEATS * 2);
  _Float16* w2h  = (_Float16*)carve((size_t)FEATS * FEATS * 2);
  _Float16* wd1h = (_Float16*)carve((size_t)FEATS * FEATS * 2);
  _Float16* wd2h = (_Float16*)carve((size_t)FEATS * OUTF * 2);
  float*    pool = (float*)   carve((size_t)NGR * FEATS * 4);
  float*    cnt  = (float*)   carve((size_t)NGR * 4);
  _Float16* hgH  = (_Float16*)carve((size_t)NGR * FEATS * 2);
  _Float16* tH   = (_Float16*)carve((size_t)NGR * FEATS * 2);

  const int B = 256;
  auto grid = [](long long n) { return (unsigned)((n + 255) / 256); };

  // --- init scratch accumulators (graph-replay safe: done every call)
  k_zero<<<grid(NND), B, 0, stream>>>(deg, NND);
  k_zero<<<grid((long long)NND * FEATS), B, 0, stream>>>(agg, NND * FEATS);
  k_zero<<<grid(NGR * FEATS), B, 0, stream>>>(pool, NGR * FEATS);
  k_zero<<<grid(NGR), B, 0, stream>>>(cnt, NGR);

  // --- weights -> f16, transposed to [N][K]
  k_convw<<<grid(FEATS * FEATS), B, 0, stream>>>(W1,  w1h,  FEATS, FEATS);
  k_convw<<<grid(FEATS * FEATS), B, 0, stream>>>(W2,  w2h,  FEATS, FEATS);
  k_convw<<<grid(FEATS * FEATS), B, 0, stream>>>(Wd1, wd1h, FEATS, FEATS);
  k_convw<<<grid(FEATS * OUTF),  B, 0, stream>>>(Wd2, wd2h, FEATS, OUTF);

  // --- in-degrees (same for both layers)
  k_degree<<<grid(NED), B, 0, stream>>>(dst, deg, NED);

  // --- layer 1: scatter, prep, WMMA GEMM + ReLU -> h1 (f32)
  k_scatter<<<grid((long long)NED * 32), B, 0, stream>>>(feat, src, dst, agg, NED);
  k_prep<<<grid((long long)NND * 32), B, 0, stream>>>(feat, agg, deg, aH);
  {
    long long threads = (long long)(NND / 16) * 32;   // one wave per 16-row tile
    k_gemm<true, false, false, FEATS, FEATS><<<grid(threads), B, 0, stream>>>(
        aH, w1h, b1, h1, nullptr, nullptr, nullptr, NND);
  }

  // --- layer 2: re-zero agg, scatter from h1, prep, GEMM + ReLU fused with pooling
  k_zero<<<grid((long long)NND * FEATS), B, 0, stream>>>(agg, NND * FEATS);
  k_scatter<<<grid((long long)NED * 32), B, 0, stream>>>(h1, src, dst, agg, NED);
  k_prep<<<grid((long long)NND * 32), B, 0, stream>>>(h1, agg, deg, aH);
  k_count<<<grid(NND), B, 0, stream>>>(gid, cnt);
  {
    long long threads = (long long)(NND / 16) * 32;
    k_gemm<true, true, false, FEATS, FEATS><<<grid(threads), B, 0, stream>>>(
        aH, w2h, b2, nullptr, nullptr, gid, pool, NND);
  }

  // --- per-graph mean -> d_out[0:8192] (hg) and f16 copy for the decoder
  k_finalize<<<grid(NGR * FEATS), B, 0, stream>>>(pool, cnt, out, hgH);

  // --- decoder: relu(hg @ Wd1 + bd1) @ Wd2 + bd2 -> d_out[8192:12288]
  k_gemm<true, false, true, FEATS, FEATS><<<1, 128, 0, stream>>>(
      hgH, wd1h, bd1, nullptr, tH, nullptr, nullptr, NGR);
  k_gemm<false, false, false, OUTF, FEATS><<<1, 128, 0, stream>>>(
      tH, wd2h, bd2, out + NGR * FEATS, nullptr, nullptr, nullptr, NGR);
}
